// Network_3066606649764
// MI455X (gfx1250) — matrix operand order repaired and checked by reading, not compiled, not run on hardware
//
#include <hip/hip_runtime.h>

typedef __attribute__((ext_vector_type(16))) __bf16 v16bf;
typedef __attribute__((ext_vector_type(8)))  __bf16 v8bf;
typedef __attribute__((ext_vector_type(8)))  float  v8f;

#define NB      16384      // batch
#define NFEAT   69         // sum(FG)
#define NEMB    24
#define XPAD    288        // 264 padded to mult of 32
#define BN_EPS  1e-5f

// group id per feature, FG = [20,20,10,10,2,2,2,1,1,1]; padded to 80
__constant__ int c_group[80] = {
  0,0,0,0,0,0,0,0,0,0,0,0,0,0,0,0,0,0,0,0,
  1,1,1,1,1,1,1,1,1,1,1,1,1,1,1,1,1,1,1,1,
  2,2,2,2,2,2,2,2,2,2,
  3,3,3,3,3,3,3,3,3,3,
  4,4, 5,5, 6,6, 7, 8, 9,
  0,0,0,0,0,0,0,0,0,0,0};

// ---------------------------------------------------------------------------
// weight prep: f32 (N,K) -> zero-padded bf16 (Npad,Kpad)
// ---------------------------------------------------------------------------
__global__ void pad_w_bf16(const float* __restrict__ src, __bf16* __restrict__ dst,
                           int N, int K, int Npad, int Kpad) {
  int t = blockIdx.x * blockDim.x + threadIdx.x;
  if (t >= Npad * Kpad) return;
  int n = t / Kpad, k = t - n * Kpad;
  dst[t] = (n < N && k < K) ? (__bf16)src[n * K + k] : (__bf16)0.f;
}

__global__ void pad_w1_f32(const float* __restrict__ src, float* __restrict__ dst) {
  int t = threadIdx.x;
  if (t < 48) dst[t] = (t < 36) ? src[t] : 0.f;
}

// ---------------------------------------------------------------------------
// Fused attention-unit + masked window pooling.  One wave (32 lanes) per row.
// Produces X[b][0..239]=pooled, [240..263]=label emb, [264..287]=0  (bf16)
// ---------------------------------------------------------------------------
__global__ __launch_bounds__(128) void fuse_attn_pool(
    const int* __restrict__ batch_item, const int* __restrict__ batch_labels,
    const float* __restrict__ node_emb,
    const __bf16* __restrict__ w0p,   // [48][96] bf16, row = out-neuron
    const float* __restrict__ b0,     // [36]
    const float* __restrict__ a0p,    // [1]
    const float* __restrict__ w1p,    // [48] f32 (padded au_w1)
    const float* __restrict__ b1p, const float* __restrict__ a1p,
    __bf16* __restrict__ X)           // [NB][XPAD]
{
  const int wid  = threadIdx.x >> 5;
  const int lane = threadIdx.x & 31;
  const int b    = blockIdx.x * 4 + wid;

  __shared__ __align__(32) __bf16 s_au[4][16 * 96];   // au_in tile, K padded to 96
  __shared__ float s_item[4][16 * 24];
  __shared__ float s_lab[4][24];
  __shared__ float s_w[4][16];
  __shared__ float s_mask[4][16];

  __bf16* au   = s_au[wid];
  float*  item = s_item[wid];
  float*  lab  = s_lab[wid];
  float*  wl   = s_w[wid];
  float*  mk   = s_mask[wid];

  const int li = batch_labels[b];
  if (lane < 24) lab[lane] = node_emb[(long)li * NEMB + lane];
  __syncthreads();

  const float a0 = a0p[0];
  const float a1 = a1p[0];
  const float b1 = b1p[0];

  // --- preload layer-0 B fragments (K-major contiguous in padded weights) ---
  const int nlo  = lane & 15;
  const int half = lane >> 4;
  v16bf bfrag[3][3];
#pragma unroll
  for (int nt = 0; nt < 3; ++nt)
#pragma unroll
    for (int kc = 0; kc < 3; ++kc)
      bfrag[nt][kc] = *(const v16bf*)(w0p + (nt * 16 + nlo) * 96 + kc * 32 + half * 16);

  float b0v[3], w1v[3];
#pragma unroll
  for (int nt = 0; nt < 3; ++nt) {
    int n = nt * 16 + nlo;
    b0v[nt] = (n < 36) ? b0[n] : 0.f;
    w1v[nt] = w1p[n];                       // zero-padded beyond 36
  }

  float acc[10];                            // per-lane (lane = emb channel) window sums
#pragma unroll
  for (int g = 0; g < 10; ++g) acc[g] = 0.f;

  for (int t = 0; t < 5; ++t) {             // 5 M-tiles of 16 features (69 -> 80)
    // ---- build au_in tile in LDS (bf16) + keep f32 item copy ----
    for (int p = lane; p < 16 * 24; p += 32) {
      int m = p / 24, e = p - m * 24;
      int f = t * 16 + m;
      int raw = (f < NFEAT) ? batch_item[(long)b * NFEAT + f] : -1;
      int idx = raw < 0 ? 0 : raw;
      float v  = node_emb[(long)idx * NEMB + e];
      float la = lab[e];
      item[m * 24 + e]   = v;
      au[m * 96 + e]       = (__bf16)v;
      au[m * 96 + 24 + e]  = (__bf16)(v * la);
      au[m * 96 + 48 + e]  = (__bf16)la;
      au[m * 96 + 72 + e]  = (__bf16)0.f;   // K pad 72..95
    }
    if (lane < 16) {
      int f = t * 16 + lane;
      int raw = (f < NFEAT) ? batch_item[(long)b * NFEAT + f] : -1;
      mk[lane] = (raw >= 0) ? 1.f : 0.f;
    }
    __syncthreads();

    // ---- A fragments from LDS (16-bit A 16x32 layout) ----
    const int row = lane & 15;
    const int kb  = half * 16;
    v16bf afrag[3];
#pragma unroll
    for (int kc = 0; kc < 3; ++kc) {
      v8bf lo = *(const v8bf*)(au + row * 96 + kc * 32 + kb);
      v8bf hi = *(const v8bf*)(au + row * 96 + kc * 32 + kb + 8);
      afrag[kc] = __builtin_shufflevector(lo, hi, 0,1,2,3,4,5,6,7,8,9,10,11,12,13,14,15);
    }

    // ---- h = PReLU(au @ w0^T + b0);  s = h @ w1 (partial, per lane) ----
    float sacc[8];
#pragma unroll
    for (int r = 0; r < 8; ++r) sacc[r] = 0.f;
#pragma unroll
    for (int nt = 0; nt < 3; ++nt) {
      v8f cc;
#pragma unroll
      for (int r = 0; r < 8; ++r) cc[r] = b0v[nt];
#pragma unroll
      for (int kc = 0; kc < 3; ++kc)
        cc = __builtin_amdgcn_wmma_f32_16x16x32_bf16(false, afrag[kc], false,
                                                     bfrag[nt][kc], (short)0, cc,
                                                     false, false);
#pragma unroll
      for (int r = 0; r < 8; ++r) {
        float h = cc[r];
        h = (h >= 0.f) ? h : a0 * h;
        sacc[r] += h * w1v[nt];
      }
    }
    // reduce over the 16 lanes of each half (n dimension)
#pragma unroll
    for (int off = 1; off < 16; off <<= 1)
#pragma unroll
      for (int r = 0; r < 8; ++r) sacc[r] += __shfl_xor(sacc[r], off, 32);

    if ((lane & 15) == 0) {                 // lane 0 -> m 0..7, lane 16 -> m 8..15
      int mb = half * 8;
#pragma unroll
      for (int r = 0; r < 8; ++r) {
        float wv = sacc[r] + b1;
        wv = (wv >= 0.f) ? wv : a1 * wv;
        wl[mb + r] = wv * mk[mb + r];       // fold in mask
      }
    }
    __syncthreads();

    // ---- weighted window pooling (lane = emb channel) ----
    if (lane < 24) {
#pragma unroll
      for (int m = 0; m < 16; ++m) {
        int f = t * 16 + m;
        if (f < NFEAT) acc[c_group[f]] += item[m * 24 + lane] * wl[m];
      }
    }
    __syncthreads();                        // tile LDS reused next iteration
  }

  const long xo = (long)b * XPAD;
  if (lane < 24) {
#pragma unroll
    for (int g = 0; g < 10; ++g) X[xo + g * 24 + lane] = (__bf16)acc[g];
    X[xo + 240 + lane] = (__bf16)lab[lane];
    X[xo + 264 + lane] = (__bf16)0.f;
  }
}

// ---------------------------------------------------------------------------
// Generic WMMA GEMM: H[M][Npad] = A[M][Kpad](bf16) @ W[Npad][Kpad]^T + bias
// One wave per 16x16 output tile.
// ---------------------------------------------------------------------------
__global__ __launch_bounds__(128) void gemm_bias_bf16(
    const __bf16* __restrict__ A, const __bf16* __restrict__ W,
    const float* __restrict__ bias, float* __restrict__ H,
    int Mtiles, int Ntiles, int Kpad, int Nreal)
{
  const int wid  = threadIdx.x >> 5;
  const int lane = threadIdx.x & 31;
  const int gid  = blockIdx.x * 4 + wid;
  if (gid >= Mtiles * Ntiles) return;       // wave-uniform
  const int mt = gid / Ntiles;
  const int nt = gid - mt * Ntiles;
  const int r    = lane & 15;
  const int half = lane >> 4;

  const __bf16* Ap = A + (long)(mt * 16 + r) * Kpad + half * 16;
  const __bf16* Wp = W + (long)(nt * 16 + r) * Kpad + half * 16;
  const int n = nt * 16 + r;

  v8f cc;
  float binit = (n < Nreal) ? bias[n] : 0.f;
#pragma unroll
  for (int i = 0; i < 8; ++i) cc[i] = binit;

  const int nk = Kpad >> 5;
  for (int kc = 0; kc < nk; ++kc) {
    v8bf lo = *(const v8bf*)(Ap + kc * 32);
    v8bf hi = *(const v8bf*)(Ap + kc * 32 + 8);
    v16bf a = __builtin_shufflevector(lo, hi, 0,1,2,3,4,5,6,7,8,9,10,11,12,13,14,15);
    v16bf w = *(const v16bf*)(Wp + kc * 32);
    cc = __builtin_amdgcn_wmma_f32_16x16x32_bf16(false, a, false, w, (short)0, cc,
                                                 false, false);
  }

  const int Npad = Ntiles * 16;
  const long base = (long)(mt * 16 + half * 8) * Npad + nt * 16 + r;
#pragma unroll
  for (int i = 0; i < 8; ++i) H[base + (long)i * Npad] = cc[i];
}

// ---------------------------------------------------------------------------
// batchnorm stats over batch axis: one block per column
// ---------------------------------------------------------------------------
__global__ __launch_bounds__(256) void bn_stats(const float* __restrict__ H,
                                                int M, int Npad,
                                                float* __restrict__ mean,
                                                float* __restrict__ var) {
  const int j = blockIdx.x;
  float s = 0.f, s2 = 0.f;
  for (int i = threadIdx.x; i < M; i += 256) {
    float v = H[(long)i * Npad + j];
    s += v; s2 += v * v;
  }
  __shared__ float rs[256], rs2[256];
  rs[threadIdx.x] = s; rs2[threadIdx.x] = s2;
  __syncthreads();
  for (int o = 128; o > 0; o >>= 1) {
    if (threadIdx.x < o) { rs[threadIdx.x] += rs[threadIdx.x + o];
                           rs2[threadIdx.x] += rs2[threadIdx.x + o]; }
    __syncthreads();
  }
  if (threadIdx.x == 0) {
    float m = rs[0] / (float)M;
    mean[j] = m;
    var[j]  = rs2[0] / (float)M - m * m;
  }
}

__global__ void bn_apply_bf16(const float* __restrict__ H,
                              const float* __restrict__ mean, const float* __restrict__ var,
                              const float* __restrict__ g, const float* __restrict__ be,
                              const float* __restrict__ ap,
                              __bf16* __restrict__ Out, int M, int Npad, int Nreal) {
  long t = (long)blockIdx.x * blockDim.x + threadIdx.x;
  if (t >= (long)M * Npad) return;
  int j = (int)(t % Npad);
  float r = 0.f;
  if (j < Nreal) {
    float v = H[t];
    v = (v - mean[j]) * rsqrtf(var[j] + BN_EPS) * g[j] + be[j];
    float a = ap[0];
    r = (v >= 0.f) ? v : a * v;
  }
  Out[t] = (__bf16)r;
}

__global__ void bn_apply_final(const float* __restrict__ H,
                               const float* __restrict__ mean, const float* __restrict__ var,
                               const float* __restrict__ g, const float* __restrict__ be,
                               const float* __restrict__ ap,
                               float* __restrict__ out, int M, int Npad) {
  int i = blockIdx.x * blockDim.x + threadIdx.x;
  if (i >= M) return;
  float v = H[(long)i * Npad];
  v = (v - mean[0]) * rsqrtf(var[0] + BN_EPS) * g[0] + be[0];
  float a = ap[0];
  out[i] = (v >= 0.f) ? v : a * v;
}

// ---------------------------------------------------------------------------
extern "C" void kernel_launch(void* const* d_in, const int* in_sizes, int n_in,
                              void* d_out, int out_size, void* d_ws, size_t ws_size,
                              hipStream_t stream) {
  const int*   batch_item   = (const int*)d_in[0];
  const int*   batch_labels = (const int*)d_in[1];
  const float* node_emb     = (const float*)d_in[2];
  // d_in[3] window_matrix: fixed 0/1 groups, encoded in c_group
  const float* au_w0 = (const float*)d_in[4];
  const float* au_b0 = (const float*)d_in[5];
  const float* au_a0 = (const float*)d_in[6];
  const float* au_w1 = (const float*)d_in[7];
  const float* au_b1 = (const float*)d_in[8];
  const float* au_a1 = (const float*)d_in[9];
  const float *lp_w[4], *lp_b[4], *lp_g[4], *lp_be[4], *lp_a[4];
  for (int i = 0; i < 4; ++i) {
    lp_w[i]  = (const float*)d_in[10 + 5 * i];
    lp_b[i]  = (const float*)d_in[11 + 5 * i];
    lp_g[i]  = (const float*)d_in[12 + 5 * i];
    lp_be[i] = (const float*)d_in[13 + 5 * i];
    lp_a[i]  = (const float*)d_in[14 + 5 * i];
  }
  float* out = (float*)d_out;

  // workspace carve-out (256B aligned)
  char* ws = (char*)d_ws;
  size_t off = 0;
  auto carve = [&](size_t bytes) -> void* {
    void* p = ws + off;
    off = (off + bytes + 255) & ~(size_t)255;
    return p;
  };
  __bf16* au_w0p = (__bf16*)carve(48 * 96 * 2);
  float*  au_w1p = (float*) carve(48 * 4);
  __bf16* w0p    = (__bf16*)carve(128 * 288 * 2);
  __bf16* w1p    = (__bf16*)carve(64 * 128 * 2);
  __bf16* w2p    = (__bf16*)carve(32 * 64 * 2);
  __bf16* w3p    = (__bf16*)carve(16 * 32 * 2);
  float*  meanB  = (float*) carve(128 * 4);
  float*  varB   = (float*) carve(128 * 4);
  __bf16* Xb     = (__bf16*)carve((size_t)NB * XPAD * 2);
  float*  Hbuf   = (float*) carve((size_t)NB * 128 * 4);
  __bf16* Abuf   = (__bf16*)carve((size_t)NB * 128 * 2);

  // ---- weight prep ----
  auto padw = [&](const float* s, __bf16* d, int N, int K, int Np, int Kp) {
    int tot = Np * Kp;
    pad_w_bf16<<<(tot + 255) / 256, 256, 0, stream>>>(s, d, N, K, Np, Kp);
  };
  padw(au_w0, au_w0p, 36, 72, 48, 96);
  padw(lp_w[0], w0p, 128, 264, 128, 288);
  padw(lp_w[1], w1p, 64, 128, 64, 128);
  padw(lp_w[2], w2p, 24, 64, 32, 64);
  padw(lp_w[3], w3p, 1, 24, 16, 32);
  pad_w1_f32<<<1, 64, 0, stream>>>(au_w1, au_w1p);

  // ---- fused attention + pooling -> X ----
  fuse_attn_pool<<<NB / 4, 128, 0, stream>>>(batch_item, batch_labels, node_emb,
                                             au_w0p, au_b0, au_a0, au_w1p, au_b1,
                                             au_a1, Xb);

  const int Mt = NB / 16;   // 1024
  auto gemm = [&](const __bf16* A, const __bf16* W, const float* bias, float* H,
                  int Ntiles, int Kpad, int Nreal) {
    int waves = Mt * Ntiles;
    gemm_bias_bf16<<<(waves + 3) / 4, 128, 0, stream>>>(A, W, bias, H, Mt, Ntiles,
                                                        Kpad, Nreal);
  };
  auto stats = [&](const float* H, int Npad, int Nreal) {
    bn_stats<<<Nreal, 256, 0, stream>>>(H, NB, Npad, meanB, varB);
  };
  auto apply = [&](const float* H, int layer, __bf16* A, int Npad, int Nreal) {
    long tot = (long)NB * Npad;
    bn_apply_bf16<<<(int)((tot + 255) / 256), 256, 0, stream>>>(
        H, meanB, varB, lp_g[layer], lp_be[layer], lp_a[layer], A, NB, Npad, Nreal);
  };

  // L0: 264(->288) -> 128
  gemm(Xb, w0p, lp_b[0], Hbuf, 8, 288, 128);
  stats(Hbuf, 128, 128);
  apply(Hbuf, 0, Abuf, 128, 128);
  // L1: 128 -> 64
  gemm(Abuf, w1p, lp_b[1], Hbuf, 4, 128, 64);
  stats(Hbuf, 64, 64);
  apply(Hbuf, 1, Abuf, 64, 64);
  // L2: 64 -> 24(->32)
  gemm(Abuf, w2p, lp_b[2], Hbuf, 2, 64, 24);
  stats(Hbuf, 32, 24);
  apply(Hbuf, 2, Abuf, 32, 24);
  // L3: 24(->32) -> 1(->16)
  gemm(Abuf, w3p, lp_b[3], Hbuf, 1, 32, 1);
  bn_stats<<<1, 256, 0, stream>>>(Hbuf, NB, 16, meanB, varB);
  bn_apply_final<<<(NB + 255) / 256, 256, 0, stream>>>(Hbuf, meanB, varB, lp_g[3],
                                                       lp_be[3], lp_a[3], out, NB, 16);
}
